// GPTImpl_48541720380014
// MI455X (gfx1250) — matrix operand order repaired and checked by reading, not compiled, not run on hardware
//
#include <hip/hip_runtime.h>
#include <cstdint>
#include <cstddef>

// ---------------------------------------------------------------------------
// GPT-2 small forward on gfx1250 (MI455X).
// - All GEMMs: bf16 operands, v_wmma_f32_16x16x32_bf16, f32 accumulate.
// - GEMM tiles staged into LDS by the Tensor Data Mover (tensor_load_to_lds,
//   TENSORcnt), double-buffered, with D#-generated padded LDS rows.
// - Weights converted f32 -> bf16 once per launch (halves weight HBM traffic).
// - Flash attention: one wave per 16-row q tile, online softmax, WMMA for
//   both Q.K^T and P.V (V pre-transposed to bf16).
// ---------------------------------------------------------------------------

typedef __attribute__((ext_vector_type(16))) __bf16        v16bf;
typedef __attribute__((ext_vector_type(8)))  __bf16        v8bf;
typedef __attribute__((ext_vector_type(8)))  float         v8f;
typedef __attribute__((ext_vector_type(4)))  unsigned int  u32x4;
typedef __attribute__((ext_vector_type(8)))  unsigned int  u32x8;

#define NB     4
#define TSEQ   1024
#define CEMB   768
#define NHEAD  12
#define HDIM   64
#define ROWS   (NB * TSEQ)        // 4096
#define VOCAB  50257
#define QKVSTR (3 * CEMB)         // 2304 (halves)
#define LDT    40                 // LDS row stride in halves (80B rows)
#define PLD    40

// ----------------------------- helpers -------------------------------------

static __device__ __forceinline__ v16bf combine16(v8bf a, v8bf b) {
  return __builtin_shufflevector(a, b, 0,1,2,3,4,5,6,7,8,9,10,11,12,13,14,15);
}

static __device__ __forceinline__ v16bf ld16(const __bf16* p0, const __bf16* p1) {
  return combine16(*(const v8bf*)p0, *(const v8bf*)p1);
}

static __device__ __forceinline__ v8f wmma_bf16(v16bf a, v16bf b, v8f c) {
  // (neg_a, A, neg_b, B, c_mod, C, reuse_a, reuse_b)
  return __builtin_amdgcn_wmma_f32_16x16x32_bf16(false, a, false, b, (short)0, c,
                                                 false, false);
}

static __device__ __forceinline__ float gelu_exact(float x) {
  return 0.5f * x * (1.0f + erff(x * 0.70710678118654752f));
}

// --- Tensor Data Mover: 2D bf16 tile (tile_rows x 32 elems) -> padded LDS ---
// D# per CDNA5 ISA ch.8: group0 = {count/lds_addr/global_addr/type},
// group1 = {data_size, pad cfg, tensor dims, tile dims, stride}.
// Pad: after every 64B row (16 DWORDs, code 3) insert 16B (4 DWORDs, code 3)
// -> LDS row stride 80B == LDT halves. Rows past tensor_dim1 read as zero.

static __device__ __forceinline__ void tdm_tile_load(
    unsigned lds_byte, const __bf16* gsrc, unsigned Kdim,
    unsigned rows_avail, unsigned tile_rows)
{
  unsigned long long ga = (unsigned long long)(const void*)gsrc;
  u32x4 g0;
  g0[0] = 1u;                                           // count=1 (valid)
  g0[1] = lds_byte;                                     // lds_addr (bytes)
  g0[2] = (unsigned)(ga & 0xffffffffu);                 // global_addr[31:0]
  g0[3] = (unsigned)((ga >> 32) & 0x01ffffffu) | (2u << 30); // addr[56:32], type=2
  u32x8 g1;
  g1[0] = (1u << 16)                                    // data_size = 2 bytes
        | (1u << 20)                                    // pad_enable
        | (3u << 22)                                    // pad_interval: 16 DWORDs
        | (3u << 25);                                   // pad_amount: 4 DWORDs
  g1[1] = (Kdim & 0xffffu) << 16;                       // tensor_dim0[15:0]
  g1[2] = (Kdim >> 16) | ((rows_avail & 0xffffu) << 16);// dim0[31:16], dim1[15:0]
  g1[3] = (rows_avail >> 16) | (32u << 16);             // dim1[31:16], tile_dim0=32
  g1[4] = tile_rows & 0xffffu;                          // tile_dim1 (tile_dim2=0)
  g1[5] = Kdim;                                         // tensor_dim0_stride[31:0]
  g1[6] = 0u;                                           // stride0[47:32], stride1 lo
  g1[7] = 0u;
  u32x4 gz = {0u, 0u, 0u, 0u};
  asm volatile("tensor_load_to_lds %0, %1, %2, %3"
               :
               : "s"(g0), "s"(g1), "s"(gz), "s"(gz)
               : "memory");
}

// ----------------------------- f32 -> bf16 copy -----------------------------

__global__ __launch_bounds__(256)
void cvtw_k(const float* __restrict__ src, __bf16* __restrict__ dst, long n) {
  long i = (long)blockIdx.x * 256 + threadIdx.x;
  long stride = (long)gridDim.x * 256;
  for (; i < n; i += stride) dst[i] = (__bf16)src[i];
}

// ----------------------------- embedding ------------------------------------

__global__ __launch_bounds__(256)
void embed_k(const int* __restrict__ idx, const float* __restrict__ wte,
             const float* __restrict__ wpe, float* __restrict__ x) {
  int row  = blockIdx.x;                 // b*T + t
  int tpos = row & (TSEQ - 1);
  size_t tok = (size_t)idx[row];
  const float* we = wte + tok * CEMB;
  const float* wp = wpe + (size_t)tpos * CEMB;
  float* xr = x + (size_t)row * CEMB;
  for (int c = threadIdx.x; c < CEMB; c += 256) xr[c] = we[c] + wp[c];
}

// ----------------------------- layernorm (f32 in, bf16 out) -----------------

__global__ __launch_bounds__(256)
void lnorm_k(const float* __restrict__ x, const float* __restrict__ w,
             __bf16* __restrict__ y) {
  __shared__ float s1[256], s2[256];
  int row = blockIdx.x, t = threadIdx.x;
  const float* xr = x + (size_t)row * CEMB;
  float a = 0.f, b = 0.f;
  for (int c = t; c < CEMB; c += 256) { float v = xr[c]; a += v; b += v * v; }
  s1[t] = a; s2[t] = b; __syncthreads();
  for (int off = 128; off > 0; off >>= 1) {
    if (t < off) { s1[t] += s1[t + off]; s2[t] += s2[t + off]; }
    __syncthreads();
  }
  float mean = s1[0] * (1.0f / CEMB);
  float var  = s2[0] * (1.0f / CEMB) - mean * mean;
  float inv  = rsqrtf(var + 1e-5f);
  __bf16* yr = y + (size_t)row * CEMB;
  for (int c = t; c < CEMB; c += 256)
    yr[c] = (__bf16)((xr[c] - mean) * inv * w[c]);
}

// ----------------------------- bf16 WMMA GEMM (TDM-fed) ---------------------
// C[M,N] = A[M,K] * W[N,K]^T (+ epilogue). M % 128 == 0, K % 32 == 0.
// Block 256 = 8 waves; macro-tile 128x64; wave -> 16 rows x 64 cols (4 accs).
// A/B tiles DMA'd to LDS by wave 0 via tensor_load_to_lds, double-buffered.
// EPI: 0 = f32 store, 1 = f32 + residual, 2 = bf16 gelu, 3 = bf16 store.

template<int EPI>
__global__ __launch_bounds__(256)
void gemm_k(const __bf16* __restrict__ A, const __bf16* __restrict__ W,
            void* __restrict__ Cout, const float* __restrict__ res,
            int M, int N, int K) {
  __shared__ __attribute__((aligned(16))) __bf16 As[2][128 * LDT];
  __shared__ __attribute__((aligned(16))) __bf16 Bs[2][64 * LDT];

  int t    = threadIdx.x;
  int lane = t & 31, wid = t >> 5;
  int g    = lane >> 4, li = lane & 15;
  int m0   = blockIdx.y * 128, n0 = blockIdx.x * 64;

  v8f acc[4];
  #pragma unroll
  for (int s = 0; s < 4; ++s) {
    #pragma unroll
    for (int r = 0; r < 8; ++r) acc[s][r] = 0.f;
  }

  unsigned ldsA[2] = { (unsigned)(size_t)&As[0][0], (unsigned)(size_t)&As[1][0] };
  unsigned ldsB[2] = { (unsigned)(size_t)&Bs[0][0], (unsigned)(size_t)&Bs[1][0] };
  const __bf16* Abase = A + (size_t)m0 * K;
  const __bf16* Wbase = W + (size_t)n0 * K;
  unsigned rowsB = (unsigned)(N - n0);    // < 64 on vocab edge: TDM zero-fills

  // prologue: fill buffer 0
  if (wid == 0) {
    tdm_tile_load(ldsA[0], Abase, (unsigned)K, 128u, 128u);
    tdm_tile_load(ldsB[0], Wbase, (unsigned)K, rowsB, 64u);
    __builtin_amdgcn_s_wait_tensorcnt(0);
  }
  __syncthreads();

  int cur = 0;
  for (int kb = 0; kb < K; kb += 32) {
    int nxt = cur ^ 1;
    bool more = (kb + 32) < K;
    if (wid == 0 && more) {               // overlap DMA with compute
      tdm_tile_load(ldsA[nxt], Abase + kb + 32, (unsigned)K, 128u, 128u);
      tdm_tile_load(ldsB[nxt], Wbase + kb + 32, (unsigned)K, rowsB, 64u);
    }

    // A fragment (16x32) for this wave's 16 rows
    const __bf16* arow = &As[cur][(wid * 16 + li) * LDT];
    v16bf af = ld16(arow + 16 * g, arow + 16 * g + 8);
    // 4 B fragments (32x16 each) spanning 64 columns
    #pragma unroll
    for (int s = 0; s < 4; ++s) {
      const __bf16* br = &Bs[cur][(s * 16 + li) * LDT];
      v16bf bf = ld16(br + 16 * g, br + 16 * g + 8);
      acc[s] = wmma_bf16(af, bf, acc[s]);
    }

    __syncthreads();                      // everyone done reading buf[cur]
    if (wid == 0 && more) __builtin_amdgcn_s_wait_tensorcnt(0);
    __syncthreads();                      // buf[nxt] ready for all waves
    cur = nxt;
  }

  // epilogue (C/D layout: VGPR r -> row r + 8*g, lane%16 -> col)
  int rbase = m0 + wid * 16 + 8 * g;
  #pragma unroll
  for (int s = 0; s < 4; ++s) {
    int col = n0 + s * 16 + li;
    if (col < N) {
      #pragma unroll
      for (int r = 0; r < 8; ++r) {
        size_t off = (size_t)(rbase + r) * N + col;
        float v = acc[s][r];
        if (EPI == 0) ((float*)Cout)[off] = v;
        if (EPI == 1) ((float*)Cout)[off] = v + res[off];
        if (EPI == 2) ((__bf16*)Cout)[off] = (__bf16)gelu_exact(v);
        if (EPI == 3) ((__bf16*)Cout)[off] = (__bf16)v;
      }
    }
  }
}

// ----------------------------- V transpose (bf16 -> bf16) -------------------
// vT[(b*12+h)*64 + d][t] = qkv[(b*T+t)*2304 + 1536 + h*64 + d]

__global__ __launch_bounds__(256)
void vtrans_k(const __bf16* __restrict__ qkv, __bf16* __restrict__ vT) {
  size_t i = (size_t)blockIdx.x * 256 + threadIdx.x;   // b,h,d,t (t fastest)
  int tt = (int)(i & (TSEQ - 1));
  size_t rest = i >> 10;
  int d = (int)(rest & (HDIM - 1)); rest >>= 6;
  int h = (int)(rest % NHEAD);
  int b = (int)(rest / NHEAD);
  vT[i] = qkv[((size_t)(b * TSEQ + tt)) * QKVSTR + 2 * CEMB + h * HDIM + d];
}

// ----------------------------- flash attention ------------------------------
// One wave per (b, h, 16-row q tile). Online softmax; WMMA for S and P.V.

__global__ __launch_bounds__(32)
void flash_k(const __bf16* __restrict__ qkv, const __bf16* __restrict__ vT,
             __bf16* __restrict__ out) {
  __shared__ __attribute__((aligned(16))) __bf16 Pl[16 * PLD];

  int qi = blockIdx.x;                    // 0..63
  int h  = blockIdx.y;                    // 0..11
  int b  = blockIdx.z;                    // 0..3
  int lane = threadIdx.x;
  int g = lane >> 4, li = lane & 15;
  int bh = b * NHEAD + h;
  int q0 = qi * 16;

  // Q fragments (A layout, d split into two 32-chunks)
  const __bf16* qrow = qkv + ((size_t)(b * TSEQ + q0 + li)) * QKVSTR + h * HDIM;
  v16bf qf0 = ld16(qrow + 16 * g,      qrow + 16 * g + 8);
  v16bf qf1 = ld16(qrow + 32 + 16 * g, qrow + 40 + 16 * g);

  float mrow[8], lrow[8];
  v8f o[4];
  #pragma unroll
  for (int r = 0; r < 8; ++r) { mrow[r] = -3.0e38f; lrow[r] = 0.f; }
  #pragma unroll
  for (int dt = 0; dt < 4; ++dt) {
    #pragma unroll
    for (int r = 0; r < 8; ++r) o[dt][r] = 0.f;
  }

  int nch = (qi >> 1) + 1;                // 32-wide k chunks (causal coverage)
  for (int jc = 0; jc < nch; ++jc) {
    int k0 = jc * 32;

    // S = Q . K^T, two 16-col sub-tiles
    v8f s0 = {0,0,0,0,0,0,0,0};
    v8f s1 = {0,0,0,0,0,0,0,0};
    const __bf16* kr0 = qkv + ((size_t)(b * TSEQ + k0 + li)) * QKVSTR + CEMB + h * HDIM;
    const __bf16* kr1 = kr0 + (size_t)16 * QKVSTR;
    #pragma unroll
    for (int dc = 0; dc < 2; ++dc) {
      v16bf bk0 = ld16(kr0 + 32 * dc + 16 * g, kr0 + 32 * dc + 16 * g + 8);
      v16bf bk1 = ld16(kr1 + 32 * dc + 16 * g, kr1 + 32 * dc + 16 * g + 8);
      v16bf qf = dc ? qf1 : qf0;
      s0 = wmma_bf16(qf, bk0, s0);
      s1 = wmma_bf16(qf, bk1, s1);
    }

    // scale + causal mask + online softmax update
    #pragma unroll
    for (int r = 0; r < 8; ++r) {
      int row = q0 + r + 8 * g;
      float a0 = (k0 + li      <= row) ? s0[r] * 0.125f : -3.0e38f;
      float a1 = (k0 + 16 + li <= row) ? s1[r] * 0.125f : -3.0e38f;
      float tm = fmaxf(a0, a1);
      #pragma unroll
      for (int off = 1; off < 16; off <<= 1)
        tm = fmaxf(tm, __shfl_xor(tm, off, 32));
      float mn = fmaxf(mrow[r], tm);
      float sc = __expf(mrow[r] - mn);
      mrow[r] = mn;
      float p0 = __expf(a0 - mn);
      float p1 = __expf(a1 - mn);
      float rs = p0 + p1;
      #pragma unroll
      for (int off = 1; off < 16; off <<= 1)
        rs += __shfl_xor(rs, off, 32);
      lrow[r] = lrow[r] * sc + rs;
      Pl[(r + 8 * g) * PLD + li]      = (__bf16)p0;
      Pl[(r + 8 * g) * PLD + 16 + li] = (__bf16)p1;
      #pragma unroll
      for (int dt = 0; dt < 4; ++dt) o[dt][r] *= sc;
    }
    __syncthreads();                      // P stores visible before frag loads

    // O += P . V  (P from LDS as A-frag, V^T rows as B-frags)
    const __bf16* prow = &Pl[li * PLD];
    v16bf pf = ld16(prow + 16 * g, prow + 16 * g + 8);
    #pragma unroll
    for (int dt = 0; dt < 4; ++dt) {
      const __bf16* vp = vT + ((size_t)(bh * HDIM + dt * 16 + li)) * TSEQ
                            + k0 + 16 * g;
      v16bf bv = ld16(vp, vp + 8);
      o[dt] = wmma_bf16(pf, bv, o[dt]);
    }
    __syncthreads();                      // done with Pl before next chunk
  }

  // normalize and store bf16 output [B*T, 768] at this head's slice
  #pragma unroll
  for (int r = 0; r < 8; ++r) {
    float inv = 1.0f / lrow[r];
    size_t rowoff = (size_t)(b * TSEQ + q0 + r + 8 * g) * CEMB + h * HDIM;
    #pragma unroll
    for (int dt = 0; dt < 4; ++dt)
      out[rowoff + dt * 16 + li] = (__bf16)(o[dt][r] * inv);
  }
}

// ----------------------------- loss ----------------------------------------

__global__ __launch_bounds__(256)
void nll_k(const float* __restrict__ logits, const int* __restrict__ tgt,
           float* __restrict__ nll) {
  __shared__ float sh[256];
  int row = blockIdx.x, t = threadIdx.x;
  const float* lr = logits + (size_t)row * VOCAB;
  float mx = -3.0e38f;
  for (int c = t; c < VOCAB; c += 256) mx = fmaxf(mx, lr[c]);
  sh[t] = mx; __syncthreads();
  for (int off = 128; off > 0; off >>= 1) {
    if (t < off) sh[t] = fmaxf(sh[t], sh[t + off]);
    __syncthreads();
  }
  mx = sh[0]; __syncthreads();
  float sm = 0.f;
  for (int c = t; c < VOCAB; c += 256) sm += expf(lr[c] - mx);
  sh[t] = sm; __syncthreads();
  for (int off = 128; off > 0; off >>= 1) {
    if (t < off) sh[t] += sh[t + off];
    __syncthreads();
  }
  if (t == 0) nll[row] = (mx + logf(sh[0])) - lr[tgt[row]];
}

__global__ __launch_bounds__(256)
void loss_k(const float* __restrict__ nll, float* __restrict__ out) {
  __shared__ float sh[256];
  int t = threadIdx.x;
  float s = 0.f;
  for (int r = t; r < ROWS; r += 256) s += nll[r];
  sh[t] = s; __syncthreads();
  for (int off = 128; off > 0; off >>= 1) {
    if (t < off) sh[t] += sh[t + off];
    __syncthreads();
  }
  if (t == 0) out[0] = sh[0] * (1.0f / ROWS);
}

// ----------------------------- launcher -------------------------------------

extern "C" void kernel_launch(void* const* d_in, const int* in_sizes, int n_in,
                              void* d_out, int out_size, void* d_ws, size_t ws_size,
                              hipStream_t stream) {
  (void)in_sizes; (void)n_in; (void)out_size; (void)ws_size;

  const int*   idx    = (const int*)d_in[0];
  const int*   tgt    = (const int*)d_in[1];
  const float* wte    = (const float*)d_in[2];
  const float* wpe    = (const float*)d_in[3];
  const float* ln1w   = (const float*)d_in[4];
  const float* wattn  = (const float*)d_in[5];
  const float* waproj = (const float*)d_in[6];
  const float* ln2w   = (const float*)d_in[7];
  const float* wfc    = (const float*)d_in[8];
  const float* wmproj = (const float*)d_in[9];
  const float* lnfw   = (const float*)d_in[10];

  float* logits = (float*)d_out;
  float* loss   = logits + (size_t)ROWS * VOCAB;

  // workspace layout
  const size_t XC = (size_t)ROWS * CEMB;          // 3,145,728
  const size_t WC = (size_t)CEMB * CEMB;          // 589,824
  float*  xA   = (float*)d_ws;                    // residual stream (f32)
  float*  xB   = xA + XC;
  float*  nllb = xB + XC;                         // ROWS
  __bf16* hbuf = (__bf16*)(nllb + ROWS);          // LN output (bf16)
  __bf16* qkvb = hbuf + XC;                       // ROWS*2304
  __bf16* attn = qkvb + 3 * XC;                   // ROWS*768
  __bf16* mlph = attn + XC;                       // ROWS*3072
  __bf16* vTb  = mlph + 4 * XC;                   // NB*NHEAD*HDIM*TSEQ
  __bf16* wab  = vTb + XC;                        // bf16 weight copies:
  __bf16* wapb = wab  + 6 * 3 * WC;               //   w_attn  6*2304*768
  __bf16* wfcb = wapb + 6 * WC;                   //   w_aproj 6*768*768
  __bf16* wmpb = wfcb + 6 * 4 * WC;               //   w_fc    6*3072*768
  __bf16* wteb = wmpb + 6 * 4 * WC;               //   w_mproj 6*768*3072
                                                  //   wte     50257*768

  dim3 blk256(256);

  // one-shot (per launch) f32 -> bf16 weight conversion
  cvtw_k<<<2048, blk256, 0, stream>>>(wattn,  wab,  (long)(6 * 3 * WC));
  cvtw_k<<<2048, blk256, 0, stream>>>(waproj, wapb, (long)(6 * WC));
  cvtw_k<<<2048, blk256, 0, stream>>>(wfc,    wfcb, (long)(6 * 4 * WC));
  cvtw_k<<<2048, blk256, 0, stream>>>(wmproj, wmpb, (long)(6 * 4 * WC));
  cvtw_k<<<2048, blk256, 0, stream>>>(wte,    wteb, (long)((size_t)VOCAB * CEMB));

  embed_k<<<ROWS, blk256, 0, stream>>>(idx, wte, wpe, xA);

  for (int l = 0; l < 6; ++l) {
    const __bf16* wa  = wab  + (size_t)l * 3 * WC;
    const __bf16* wap = wapb + (size_t)l * WC;
    const __bf16* wf  = wfcb + (size_t)l * 4 * WC;
    const __bf16* wmp = wmpb + (size_t)l * 4 * WC;

    // ln1 -> h ; qkv = h @ wa^T  (bf16 out)
    lnorm_k<<<ROWS, blk256, 0, stream>>>(xA, ln1w + l * CEMB, hbuf);
    gemm_k<3><<<dim3(QKVSTR / 64, ROWS / 128), blk256, 0, stream>>>(
        hbuf, wa, qkvb, nullptr, ROWS, QKVSTR, CEMB);

    // attention
    vtrans_k<<<(NB * NHEAD * HDIM * TSEQ) / 256, blk256, 0, stream>>>(qkvb, vTb);
    flash_k<<<dim3(TSEQ / 16, NHEAD, NB), dim3(32), 0, stream>>>(qkvb, vTb, attn);

    // xB = attn @ wap^T + xA  (f32 residual)
    gemm_k<1><<<dim3(CEMB / 64, ROWS / 128), blk256, 0, stream>>>(
        attn, wap, xB, xA, ROWS, CEMB, CEMB);

    // ln2 -> h ; mlp hidden = gelu(h @ wf^T)  (bf16 out)
    lnorm_k<<<ROWS, blk256, 0, stream>>>(xB, ln2w + l * CEMB, hbuf);
    gemm_k<2><<<dim3(4 * CEMB / 64, ROWS / 128), blk256, 0, stream>>>(
        hbuf, wf, mlph, nullptr, ROWS, 4 * CEMB, CEMB);

    // xA = mlph @ wmp^T + xB  (f32 residual)
    gemm_k<1><<<dim3(CEMB / 64, ROWS / 128), blk256, 0, stream>>>(
        mlph, wmp, xA, xB, ROWS, CEMB, 4 * CEMB);
  }

  // final LN + tied lm_head: logits = lnf(x) @ wte^T
  lnorm_k<<<ROWS, blk256, 0, stream>>>(xA, lnfw, hbuf);
  gemm_k<0><<<dim3((VOCAB + 63) / 64, ROWS / 128), blk256, 0, stream>>>(
      hbuf, wteb, logits, nullptr, ROWS, VOCAB, CEMB);

  // loss
  nll_k<<<ROWS, blk256, 0, stream>>>(logits, tgt, nllb);
  loss_k<<<1, blk256, 0, stream>>>(nllb, loss);
}
